// EnhancedTemporalMemoryBank_6382321402038
// MI455X (gfx1250) — compile-verified, matrix operand order checked by reading, not run on hardware
//
#include <hip/hip_runtime.h>
#include <hip/hip_bf16.h>

#define MSZ 65536
#define DD  256
#define HH  4
#define HD  64
#define BB  512
#define NPART 16
#define PARTM (MSZ / NPART)   // 4096 keys per partition

typedef __attribute__((ext_vector_type(16))) __bf16         v16bf;
typedef __attribute__((ext_vector_type(8)))  __bf16         v8bf;
typedef __attribute__((ext_vector_type(8)))  float          v8f;
typedef __attribute__((ext_vector_type(8)))  unsigned short ushort8;
typedef __attribute__((ext_vector_type(4)))  int            v4i;

#if defined(__gfx1250__) && __has_builtin(__builtin_amdgcn_global_load_async_to_lds_b128) && __has_builtin(__builtin_amdgcn_s_wait_asynccnt)
#define HAVE_ASYNC 1
#else
#define HAVE_ASYNC 0
#endif

// ---------- helpers ----------
__device__ inline unsigned short f2bfu(float f) {
  return __builtin_bit_cast(unsigned short, static_cast<__bf16>(f));
}

__device__ inline v8f vzero8() {
  v8f z;
#pragma unroll
  for (int i = 0; i < 8; ++i) z[i] = 0.0f;
  return z;
}

__device__ inline v8f wmma_bf16(v16bf a, v16bf b, v8f c) {
  return __builtin_amdgcn_wmma_f32_16x16x32_bf16(false, a, false, b, (short)0, c, false, false);
}

// A/B fragment (16x32) from row-major K-contiguous source.
// element e: k = 8*half + e (e<8) ; k = 8*half + 16 + (e-8) (e>=8)
__device__ inline v16bf load_frag_f32(const float* __restrict__ p, int half) {
  v8bf lo = __builtin_convertvector(*(const v8f*)(p + 8 * half), v8bf);
  v8bf hi = __builtin_convertvector(*(const v8f*)(p + 16 + 8 * half), v8bf);
  return __builtin_shufflevector(lo, hi, 0, 1, 2, 3, 4, 5, 6, 7, 8, 9, 10, 11, 12, 13, 14, 15);
}
__device__ inline v16bf load_frag_u16(const unsigned short* __restrict__ p, int half) {
  v8bf lo = __builtin_bit_cast(v8bf, *(const ushort8*)(p + 8 * half));
  v8bf hi = __builtin_bit_cast(v8bf, *(const ushort8*)(p + 16 + 8 * half));
  return __builtin_shufflevector(lo, hi, 0, 1, 2, 3, 4, 5, 6, 7, 8, 9, 10, 11, 12, 13, 14, 15);
}

__device__ inline float wave_max16(float v) {
#pragma unroll
  for (int m = 1; m < 16; m <<= 1) v = fmaxf(v, __shfl_xor(v, m, 32));
  return v;
}
__device__ inline float wave_sum16(float v) {
#pragma unroll
  for (int m = 1; m < 16; m <<= 1) v += __shfl_xor(v, m, 32);
  return v;
}

#if HAVE_ASYNC
// Stage one 32-key x 64-feature bf16 K chunk (4 KB) for head h into LDS
// via GLOBAL_LOAD_ASYNC_TO_LDS_B128 (8 per-lane 16B transfers, ASYNCcnt).
__device__ inline void issue_k_async(const unsigned short* __restrict__ Kb, int mb, int h,
                                     unsigned short* dst, int lane) {
#pragma unroll
  for (int j = 0; j < 8; ++j) {
    const int o   = j * 512 + lane * 16;  // byte offset inside 4KB blob
    const int row = o >> 7;               // 128 B per key row
    const int rb  = (o & 127) >> 1;       // u16 offset inside row
    const unsigned short* g = Kb + (size_t)(mb + row) * DD + h * HD + rb;
    __builtin_amdgcn_global_load_async_to_lds_b128(
        (__attribute__((address_space(1))) v4i*)g,
        (__attribute__((address_space(3))) v4i*)(dst + (o >> 1)), 0, 0);
  }
}
#endif

// ---------- dense GEMM: out[512,256] = A[512,256] x W[256,256]^T + bias ----------
__global__ __launch_bounds__(32) void gemm512x256_kernel(
    const float* __restrict__ A, const float* __restrict__ W,
    const float* __restrict__ bias, float* __restrict__ out) {
  const int lane = threadIdx.x & 31;
  const int half = lane >> 4, col = lane & 15;
  const int mb = blockIdx.x * 16, nb = blockIdx.y * 16;
  v8f acc = vzero8();
  for (int kk = 0; kk < DD; kk += 32) {
    v16bf af = load_frag_f32(A + (size_t)(mb + col) * DD + kk, half);
    v16bf bf = load_frag_f32(W + (size_t)(nb + col) * DD + kk, half);
    acc = wmma_bf16(af, bf, acc);
  }
  const float b = bias[nb + col];
#pragma unroll
  for (int i = 0; i < 8; ++i)
    out[(size_t)(mb + i + 8 * half) * DD + nb + col] = acc[i] + b;
}

// ---------- K = mem·Wk^T (bf16 [M,D]) and Vt = (mem·Wv^T)^T (bf16 [D,M]) ----------
__global__ __launch_bounds__(32) void proj_kv_kernel(
    const float* __restrict__ memory,
    const float* __restrict__ Wk, const float* __restrict__ bk,
    const float* __restrict__ Wv, const float* __restrict__ bv,
    unsigned short* __restrict__ Kb, unsigned short* __restrict__ Vt) {
  const int lane = threadIdx.x & 31;
  const int half = lane >> 4, col = lane & 15;
  const int mb = blockIdx.x * 16, nb = blockIdx.y * 16;
  const int which = blockIdx.z;
  const float* W    = which ? Wv : Wk;
  const float* bias = which ? bv : bk;
  v8f acc = vzero8();
  for (int kk = 0; kk < DD; kk += 32) {
    v16bf af = load_frag_f32(memory + (size_t)(mb + col) * DD + kk, half);
    v16bf bf = load_frag_f32(W + (size_t)(nb + col) * DD + kk, half);
    acc = wmma_bf16(af, bf, acc);
  }
  const float b = bias[nb + col];
  if (which == 0) {
#pragma unroll
    for (int i = 0; i < 8; ++i)
      Kb[(size_t)(mb + i + 8 * half) * DD + nb + col] = f2bfu(acc[i] + b);
  } else {
    v8f tmp;
#pragma unroll
    for (int i = 0; i < 8; ++i) tmp[i] = acc[i] + b;
    *(ushort8*)(Vt + (size_t)(nb + col) * MSZ + mb + 8 * half) =
        __builtin_bit_cast(ushort8, __builtin_convertvector(tmp, v8bf));
  }
}

// ---------- attention partials: block = (16 queries) x (1/16 of M); wave = head ----------
__global__ __launch_bounds__(128) void attn_partial_kernel(
    const float* __restrict__ Qw, const unsigned short* __restrict__ Kb,
    const unsigned short* __restrict__ Vt, float* __restrict__ partO,
    float* __restrict__ pmax, float* __restrict__ psum) {
  __shared__ __align__(16) unsigned short plds[HH][16][32];
#if HAVE_ASYNC
  __shared__ __align__(16) unsigned short kstage[HH][2][32 * HD];
#endif
  const int tid  = threadIdx.x;
  const int h    = tid >> 5;
  const int lane = tid & 31;
  const int half = lane >> 4, col = lane & 15;
  const int bblock = blockIdx.x;
  const int part   = blockIdx.y;
  const int mstart = part * PARTM;
  const float scale = 0.125f;  // 1/sqrt(64)

  v16bf qf[2];
  {
    const float* qrow = Qw + (size_t)(bblock * 16 + col) * DD + h * HD;
#pragma unroll
    for (int s = 0; s < 2; ++s) qf[s] = load_frag_f32(qrow + s * 32, half);
  }

  v8f oacc[4];
#pragma unroll
  for (int nt = 0; nt < 4; ++nt) oacc[nt] = vzero8();
  float runmax[8], runsum[8];
#pragma unroll
  for (int i = 0; i < 8; ++i) { runmax[i] = -1e30f; runsum[i] = 0.0f; }

#if HAVE_ASYNC
  issue_k_async(Kb, mstart, h, &kstage[h][0][0], lane);
#endif

  for (int mb = mstart; mb < mstart + PARTM; mb += 32) {
#if HAVE_ASYNC
    const int buf = ((mb - mstart) >> 5) & 1;
    if (mb + 32 < mstart + PARTM) {
      issue_k_async(Kb, mb + 32, h, &kstage[h][buf ^ 1][0], lane);
      __builtin_amdgcn_s_wait_asynccnt(8);  // current buf's (older) 8 retired
    } else {
      __builtin_amdgcn_s_wait_asynccnt(0);
    }
#endif
    v8f st[2];
#pragma unroll
    for (int t = 0; t < 2; ++t) {
      v8f acc = vzero8();
#if HAVE_ASYNC
      const unsigned short* krow = &kstage[h][buf][(t * 16 + col) * HD];
#else
      const unsigned short* krow = Kb + (size_t)(mb + t * 16 + col) * DD + h * HD;
#endif
#pragma unroll
      for (int s = 0; s < 2; ++s)
        acc = wmma_bf16(qf[s], load_frag_u16(krow + s * 32, half), acc);
#pragma unroll
      for (int i = 0; i < 8; ++i) acc[i] *= scale;
      st[t] = acc;
    }
    float nmax[8], fac[8], psm[8];
#pragma unroll
    for (int i = 0; i < 8; ++i) {
      float cm = wave_max16(fmaxf(st[0][i], st[1][i]));
      nmax[i] = fmaxf(runmax[i], cm);
      fac[i]  = __expf(runmax[i] - nmax[i]);
    }
#pragma unroll
    for (int i = 0; i < 8; ++i) {
      st[0][i] = __expf(st[0][i] - nmax[i]);
      st[1][i] = __expf(st[1][i] - nmax[i]);
      psm[i]   = wave_sum16(st[0][i] + st[1][i]);
      runsum[i] = runsum[i] * fac[i] + psm[i];
      runmax[i] = nmax[i];
    }
#pragma unroll
    for (int nt = 0; nt < 4; ++nt)
#pragma unroll
      for (int i = 0; i < 8; ++i) oacc[nt][i] *= fac[i];

    // P (C layout) -> LDS -> A-fragment layout (wave-internal; DScnt ordered)
#pragma unroll
    for (int t = 0; t < 2; ++t)
#pragma unroll
      for (int i = 0; i < 8; ++i)
        plds[h][i + 8 * half][t * 16 + col] = f2bfu(st[t][i]);
    v16bf pf = load_frag_u16(&plds[h][col][0], half);

#pragma unroll
    for (int nt = 0; nt < 4; ++nt) {
      const unsigned short* vrow = Vt + (size_t)(h * HD + nt * 16 + col) * MSZ + mb;
      if (mb + 32 < mstart + PARTM) __builtin_prefetch(vrow + 32, 0, 1);
      oacc[nt] = wmma_bf16(pf, load_frag_u16(vrow, half), oacc[nt]);
    }
  }

  // write UNnormalized partial O + per-(row,head) stats
#pragma unroll
  for (int nt = 0; nt < 4; ++nt)
#pragma unroll
    for (int i = 0; i < 8; ++i) {
      const int brow = bblock * 16 + i + 8 * half;
      partO[((size_t)part * BB + brow) * DD + h * HD + nt * 16 + col] = oacc[nt][i];
    }
  if (col == 0) {
#pragma unroll
    for (int i = 0; i < 8; ++i) {
      const int brow = bblock * 16 + i + 8 * half;
      pmax[((size_t)part * BB + brow) * HH + h] = runmax[i];
      psum[((size_t)part * BB + brow) * HH + h] = runsum[i];
    }
  }
}

// ---------- combine partials: log-sum-exp merge over 16 partitions ----------
__global__ __launch_bounds__(256) void attn_combine_kernel(
    const float* __restrict__ partO, const float* __restrict__ pmax,
    const float* __restrict__ psum, float* __restrict__ Ow,
    float* __restrict__ gmaxA, float* __restrict__ gsumA) {
  const int b = blockIdx.x;
  const int d = threadIdx.x;
  const int h = d >> 6;
  float gm = -1e30f;
#pragma unroll
  for (int p = 0; p < NPART; ++p)
    gm = fmaxf(gm, pmax[((size_t)p * BB + b) * HH + h]);
  float gs = 0.0f, o = 0.0f;
#pragma unroll
  for (int p = 0; p < NPART; ++p) {
    const float f = __expf(pmax[((size_t)p * BB + b) * HH + h] - gm);
    gs += psum[((size_t)p * BB + b) * HH + h] * f;
    o  += partO[((size_t)p * BB + b) * DD + d] * f;
  }
  Ow[(size_t)b * DD + d] = o / gs;
  if ((d & 63) == 0) { gmaxA[b * HH + h] = gm; gsumA[b * HH + h] = gs; }
}

// ---------- per-partition argmax of head-summed attention (needs global stats) ----------
__global__ __launch_bounds__(128) void attn_argmax_kernel(
    const float* __restrict__ Qw, const unsigned short* __restrict__ Kb,
    const float* __restrict__ gmaxA, const float* __restrict__ gsumA,
    float* __restrict__ pbestv, int* __restrict__ pbesti) {
  __shared__ float atile[16][32];
#if HAVE_ASYNC
  __shared__ __align__(16) unsigned short kstage[HH][2][32 * HD];
#endif
  const int tid  = threadIdx.x;
  const int h    = tid >> 5;
  const int lane = tid & 31;
  const int half = lane >> 4, col = lane & 15;
  const int bblock = blockIdx.x;
  const int part   = blockIdx.y;
  const int mstart = part * PARTM;
  const float scale = 0.125f;

  for (int i = tid; i < 16 * 32; i += 128) ((float*)atile)[i] = 0.0f;

  v16bf qf[2];
  float gm[8], gs[8];
  {
    const float* qrow = Qw + (size_t)(bblock * 16 + col) * DD + h * HD;
#pragma unroll
    for (int s = 0; s < 2; ++s) qf[s] = load_frag_f32(qrow + s * 32, half);
#pragma unroll
    for (int i = 0; i < 8; ++i) {
      const int brow = bblock * 16 + i + 8 * half;
      gm[i] = gmaxA[brow * HH + h];
      gs[i] = gsumA[brow * HH + h];
    }
  }
  float bestv = -1.0f;
  int   besti = mstart;
#if HAVE_ASYNC
  issue_k_async(Kb, mstart, h, &kstage[h][0][0], lane);
#endif
  __syncthreads();

  for (int mb = mstart; mb < mstart + PARTM; mb += 32) {
#if HAVE_ASYNC
    const int buf = ((mb - mstart) >> 5) & 1;
    if (mb + 32 < mstart + PARTM) {
      issue_k_async(Kb, mb + 32, h, &kstage[h][buf ^ 1][0], lane);
      __builtin_amdgcn_s_wait_asynccnt(8);
    } else {
      __builtin_amdgcn_s_wait_asynccnt(0);
    }
#endif
#pragma unroll
    for (int t = 0; t < 2; ++t) {
      v8f acc = vzero8();
#if HAVE_ASYNC
      const unsigned short* krow = &kstage[h][buf][(t * 16 + col) * HD];
#else
      const unsigned short* krow = Kb + (size_t)(mb + t * 16 + col) * DD + h * HD;
#endif
#pragma unroll
      for (int s = 0; s < 2; ++s)
        acc = wmma_bf16(qf[s], load_frag_u16(krow + s * 32, half), acc);
#pragma unroll
      for (int i = 0; i < 8; ++i) {
        const float a = __expf(acc[i] * scale - gm[i]) / gs[i];
        atomicAdd(&atile[i + 8 * half][t * 16 + col], a);
      }
    }
    __syncthreads();
    if (tid < 32) {
      const int row = lane >> 1;
      const int cb  = (lane & 1) * 16;
#pragma unroll
      for (int c = 0; c < 16; ++c) {
        const float v = atile[row][cb + c];
        if (v > bestv) { bestv = v; besti = mb + cb + c; }
      }
    }
    __syncthreads();
    for (int i = tid; i < 16 * 32; i += 128) ((float*)atile)[i] = 0.0f;
    __syncthreads();
  }
  if (tid < 32) {
    const float ov = __shfl_xor(bestv, 1, 32);
    const int   oi = __shfl_xor(besti, 1, 32);
    if (ov > bestv || (ov == bestv && oi < besti)) { bestv = ov; besti = oi; }
    if ((lane & 1) == 0) {
      const int brow = bblock * 16 + (lane >> 1);
      pbestv[brow * NPART + part] = bestv;
      pbesti[brow * NPART + part] = besti;
    }
  }
}

__global__ __launch_bounds__(512) void attn_argmax_reduce_kernel(
    const float* __restrict__ pbestv, const int* __restrict__ pbesti,
    int* __restrict__ amax) {
  const int b = threadIdx.x;
  float bv = -1.0f;
  int   bi = 0;
#pragma unroll
  for (int p = 0; p < NPART; ++p) {   // ascending p => first-index tie-break
    const float v = pbestv[b * NPART + p];
    const int   i = pbesti[b * NPART + p];
    if (v > bv || (v == bv && i < bi)) { bv = v; bi = i; }
  }
  amax[b] = bi;
}

// ---------- sequential LRU-gated memory write: single workgroup ----------
__global__ __launch_bounds__(1024) void scan_kernel(
    const float* __restrict__ query, const float* __restrict__ Wg,
    const float* __restrict__ bg, const float* __restrict__ retrieved,
    const int* __restrict__ amax, float* __restrict__ mem,
    float* __restrict__ usage) {
  __shared__ float rv[1024];
  __shared__ int   ri[1024];
  __shared__ float xv[768];
  __shared__ float gpart[1024];
  __shared__ float s_sum;
  __shared__ int   s_idx;
  const int t = threadIdx.x;

  float ls = 0.0f;
  for (int m = t; m < MSZ; m += 1024) ls += usage[m];
  rv[t] = ls;
  __syncthreads();
  for (int off = 512; off > 0; off >>= 1) {
    if (t < off) rv[t] += rv[t + off];
    __syncthreads();
  }
  if (t == 0) s_sum = rv[0];
  __syncthreads();

  for (int step = 0; step < BB; ++step) {
    float mv = 3.4e38f;
    int   mi = 0;
    for (int m = t; m < MSZ; m += 1024) {
      const float u = usage[m];
      if (u < mv) { mv = u; mi = m; }
    }
    rv[t] = mv; ri[t] = mi;
    __syncthreads();
    for (int off = 512; off > 0; off >>= 1) {
      if (t < off) {
        if (rv[t + off] < rv[t] || (rv[t + off] == rv[t] && ri[t + off] < ri[t])) {
          rv[t] = rv[t + off]; ri[t] = ri[t + off];
        }
      }
      __syncthreads();
    }
    if (t == 0) {
      const int lru = ri[0];
      const int mx  = amax[step];
      const float mean = s_sum * (1.0f / (float)MSZ);
      s_idx = (usage[mx] < mean) ? mx : lru;
    }
    __syncthreads();
    const int idx = s_idx;

    if (t < 256)      xv[t] = query[(size_t)step * DD + t];
    else if (t < 512) xv[t] = mem[(size_t)idx * DD + (t - 256)];
    else if (t < 768) xv[t] = retrieved[(size_t)step * DD + (t - 512)];
    __syncthreads();

    {
      const int j = t >> 2, part = t & 3;
      const float* w = Wg + (size_t)j * 768 + part * 192;
      const float* x = xv + part * 192;
      float acc = 0.0f;
#pragma unroll 4
      for (int k = 0; k < 192; ++k) acc += w[k] * x[k];
      gpart[t] = acc;
    }
    __syncthreads();
    if (t < 256) {
      float g = gpart[4 * t] + gpart[4 * t + 1] + gpart[4 * t + 2] + gpart[4 * t + 3] + bg[t];
      g = 1.0f / (1.0f + __expf(-g));
      const float old = xv[256 + t];
      mem[(size_t)idx * DD + t] = g * xv[t] + (1.0f - g) * old;
    }
    if (t == 0) { usage[idx] += 1.0f; s_sum += 1.0f; }
    __threadfence_block();
    __syncthreads();
  }
}

// ---------- launcher ----------
extern "C" void kernel_launch(void* const* d_in, const int* in_sizes, int n_in,
                              void* d_out, int out_size, void* d_ws, size_t ws_size,
                              hipStream_t stream) {
  const float* query  = (const float*)d_in[0];
  const float* memory = (const float*)d_in[1];
  const float* Wq = (const float*)d_in[2];
  const float* bq = (const float*)d_in[3];
  const float* Wk = (const float*)d_in[4];
  const float* bk = (const float*)d_in[5];
  const float* Wv = (const float*)d_in[6];
  const float* bv = (const float*)d_in[7];
  const float* Wo = (const float*)d_in[8];
  const float* bo = (const float*)d_in[9];
  const float* Wg = (const float*)d_in[10];
  const float* bg = (const float*)d_in[11];
  const float* usage_in = (const float*)d_in[12];

  float* out_retr  = (float*)d_out;                 // [512,256]
  float* out_mem   = out_retr + (size_t)BB * DD;    // [65536,256]
  float* out_usage = out_mem + (size_t)MSZ * DD;    // [65536]

  char* w = (char*)d_ws;
  float* Qw = (float*)w;                   w += (size_t)BB * DD * 4;
  unsigned short* Kb = (unsigned short*)w; w += (size_t)MSZ * DD * 2;
  unsigned short* Vt = (unsigned short*)w; w += (size_t)MSZ * DD * 2;
  float* Ow = (float*)w;                   w += (size_t)BB * DD * 4;
  float* partO = (float*)w;                w += (size_t)NPART * BB * DD * 4;
  float* pmax  = (float*)w;                w += (size_t)NPART * BB * HH * 4;
  float* psum  = (float*)w;                w += (size_t)NPART * BB * HH * 4;
  float* gmaxA = (float*)w;                w += (size_t)BB * HH * 4;
  float* gsumA = (float*)w;                w += (size_t)BB * HH * 4;
  float* pbestv = (float*)w;               w += (size_t)BB * NPART * 4;
  int*   pbesti = (int*)w;                 w += (size_t)BB * NPART * 4;
  int*   amax   = (int*)w;                 w += (size_t)BB * 4;

  (void)hipMemcpyAsync(out_mem, memory, (size_t)MSZ * DD * sizeof(float),
                       hipMemcpyDeviceToDevice, stream);
  (void)hipMemcpyAsync(out_usage, usage_in, (size_t)MSZ * sizeof(float),
                       hipMemcpyDeviceToDevice, stream);

  gemm512x256_kernel<<<dim3(BB / 16, DD / 16), 32, 0, stream>>>(query, Wq, bq, Qw);
  proj_kv_kernel<<<dim3(MSZ / 16, DD / 16, 2), 32, 0, stream>>>(
      memory, Wk, bk, Wv, bv, Kb, Vt);
  attn_partial_kernel<<<dim3(BB / 16, NPART), 128, 0, stream>>>(
      Qw, Kb, Vt, partO, pmax, psum);
  attn_combine_kernel<<<BB, 256, 0, stream>>>(partO, pmax, psum, Ow, gmaxA, gsumA);
  attn_argmax_kernel<<<dim3(BB / 16, NPART), 128, 0, stream>>>(
      Qw, Kb, gmaxA, gsumA, pbestv, pbesti);
  attn_argmax_reduce_kernel<<<1, BB, 0, stream>>>(pbestv, pbesti, amax);
  gemm512x256_kernel<<<dim3(BB / 16, DD / 16), 32, 0, stream>>>(Ow, Wo, bo, out_retr);
  scan_kernel<<<1, 1024, 0, stream>>>(query, Wg, bg, out_retr, amax, out_mem, out_usage);
}
